// mLSTM_71777493450985
// MI455X (gfx1250) — matrix operand order repaired and checked by reading, not compiled, not run on hardware
//
#include <hip/hip_runtime.h>
#include <hip/hip_bf16.h>

// ---------------------------------------------------------------------------
// mLSTM forward for B=8, S=256, D=H=256 on gfx1250 (MI455X).
// Phase 1: five projection GEMMs via v_wmma_f32_16x16x32_bf16 with bf16
//          hi/lo split-accumulation (near-fp32 accuracy), native bf16 cvts.
// Phase 2: sequential scan; per-batch 256x256 fp32 state C kept entirely in
//          LDS (256KB of the WGP's 320KB), one workgroup per batch element.
//          Row phase uses 16B/lane LDS vectors (ds_load_b128 pattern).
// ---------------------------------------------------------------------------

typedef __attribute__((ext_vector_type(16))) __bf16 v16bf;
typedef __attribute__((ext_vector_type(8)))  float  v8f;

#define BATCH 8
#define SEQ   256
#define DIM   256
#define NVEC  (2048 * 256)   // elements per projection matrix [B*S, H]

// ---------------------------------------------------------------------------
// Kernel 1: Y[mat] = X @ W[mat] + b[mat]   (exp() applied for mat 3,4)
// One wave per 16x16 output tile. 5 * (2048/16) * (256/16) = 10240 waves
// = 1280 blocks of 8 waves. K looped in chunks of 32 with bf16 hi/lo split:
// acc += Ahi*Bhi + Ahi*Blo + Alo*Bhi   (3 WMMAs per chunk).
// ---------------------------------------------------------------------------
__global__ __launch_bounds__(256) void mlstm_proj_kernel(
    const float* __restrict__ X,
    const float* __restrict__ Wq, const float* __restrict__ bq,
    const float* __restrict__ Wk, const float* __restrict__ bk,
    const float* __restrict__ Wv, const float* __restrict__ bv,
    const float* __restrict__ Wi, const float* __restrict__ bi,
    const float* __restrict__ Wf, const float* __restrict__ bf_,
    float* __restrict__ ws)
{
    const int tid  = threadIdx.x;
    const int lane = tid & 31;
    const int wave = tid >> 5;
    const int gw   = blockIdx.x * 8 + wave;   // global wave id, < 10240

    const int mat = gw >> 11;                 // 2048 tiles per matrix
    const int rem = gw & 2047;
    const int tm  = rem >> 4;                 // 0..127  (M tile)
    const int tn  = rem & 15;                 // 0..15   (N tile)

    const float* W;  const float* bias;
    if      (mat == 0) { W = Wq; bias = bq; }
    else if (mat == 1) { W = Wk; bias = bk; }
    else if (mat == 2) { W = Wv; bias = bv; }
    else if (mat == 3) { W = Wi; bias = bi; }
    else               { W = Wf; bias = bf_; }
    float* Y = ws + (size_t)mat * NVEC;

    const int m16  = lane & 15;
    const int half = lane >> 4;
    const int arow = tm * 16 + m16;           // A row handled by this lane
    const int bcol = tn * 16 + m16;           // B/C column handled by this lane

    v8f acc = {};

#pragma unroll 2
    for (int kb = 0; kb < 8; ++kb) {
        // ---- A fragment (16-bit A 16x32): lane holds row `arow`;
        //      elems 0..7 -> K = kb*32 + half*8 + e,
        //      elems 8..15 -> K = kb*32 + 16 + half*8 + (e-8).
        const int k0 = kb * 32 + half * 8;
        const float4 a0 = *(const float4*)(X + arow * DIM + k0);
        const float4 a1 = *(const float4*)(X + arow * DIM + k0 + 4);
        const float4 a2 = *(const float4*)(X + arow * DIM + k0 + 16);
        const float4 a3 = *(const float4*)(X + arow * DIM + k0 + 20);
        float ar[16] = { a0.x, a0.y, a0.z, a0.w,  a1.x, a1.y, a1.z, a1.w,
                         a2.x, a2.y, a2.z, a2.w,  a3.x, a3.y, a3.z, a3.w };

        // ---- B fragment (16-bit B 32x16): lane holds col `bcol`;
        //      elems 0..7 -> K = kb*32 + half*8 + e, elems 8..15 -> K = kb*32 + 16 + half*8 + (e-8)
        const int kbase = kb * 32 + half * 8;
        float br[16];
#pragma unroll
        for (int e = 0; e < 16; ++e)
            br[e] = W[(kbase + (e & 7) + ((e >> 3) << 4)) * DIM + bcol];

        v16bf ah, al, bh, bl;
#pragma unroll
        for (int e = 0; e < 16; ++e) {
            const __bf16 hA = (__bf16)ar[e];          // native RNE cvt
            ah[e] = hA;
            al[e] = (__bf16)(ar[e] - (float)hA);      // residual term
            const __bf16 hB = (__bf16)br[e];
            bh[e] = hB;
            bl[e] = (__bf16)(br[e] - (float)hB);
        }

        acc = __builtin_amdgcn_wmma_f32_16x16x32_bf16(false, ah, false, bh,
                                                      (short)0, acc, false, false);
        acc = __builtin_amdgcn_wmma_f32_16x16x32_bf16(false, ah, false, bl,
                                                      (short)0, acc, false, false);
        acc = __builtin_amdgcn_wmma_f32_16x16x32_bf16(false, al, false, bh,
                                                      (short)0, acc, false, false);
    }

    // C/D layout: VGPR r, lane l -> row = (l>>4)*8 + r, col = l&15 (tile-local)
    const float bv_ = bias[bcol];
#pragma unroll
    for (int r = 0; r < 8; ++r) {
        const int mo = tm * 16 + half * 8 + r;
        float y = acc[r] + bv_;
        if (mat >= 3) y = expf(y);            // exponential input/forget gates
        Y[(size_t)mo * DIM + bcol] = y;
    }
}

// ---------------------------------------------------------------------------
// Kernel 2: sequential scan. One block (8 waves / 256 threads) per batch.
// Dynamic LDS layout (floats):
//   C      [0,      65536)  : 256x256 state, row-major
//   hb     [65536,  65792)  : hidden state h
//   vecs   [65792,  67072)  : q,k,v,i,f step vectors (5 x 256)
//   ap     [67072,  69120)  : per-wave attn partials (8 x 256)
//   gs/bs  [69120,  69632)  : LayerNorm gamma/beta
// ---------------------------------------------------------------------------
#define SM_C   0
#define SM_H   65536
#define SM_V   65792
#define SM_AP  67072
#define SM_G   69120
#define SM_B   69376
#define SM_FLOATS 69632
#define SM_BYTES  (SM_FLOATS * 4)

__global__ __launch_bounds__(256) void mlstm_scan_kernel(
    const float* __restrict__ q,  const float* __restrict__ k,
    const float* __restrict__ v,  const float* __restrict__ ig,
    const float* __restrict__ fg,
    const float* __restrict__ Wo, const float* __restrict__ bo,
    const float* __restrict__ lng, const float* __restrict__ lnb,
    float* __restrict__ ys, float* __restrict__ hf, float* __restrict__ Cf)
{
    extern __shared__ float sm[];
    float* C   = sm + SM_C;
    float* hb  = sm + SM_H;
    float* qs  = sm + SM_V;
    float* ks  = sm + SM_V + 256;
    float* vs  = sm + SM_V + 512;
    float* is_ = sm + SM_V + 768;
    float* fs  = sm + SM_V + 1024;
    float* ap  = sm + SM_AP;
    float* gs  = sm + SM_G;
    float* bs  = sm + SM_B;

    const int tid  = threadIdx.x;
    const int lane = tid & 31;
    const int wave = tid >> 5;
    const int bb   = blockIdx.x;
    const int colb = lane * 8;                 // 8 contiguous columns per lane

    // ---- init: C = 0, h = 0, cache LN params
    {
        const float4 z4 = make_float4(0.f, 0.f, 0.f, 0.f);
        float4* C4 = (float4*)C;
        for (int i = tid; i < 16384; i += 256) C4[i] = z4;
        hb[tid] = 0.0f;
        gs[tid] = lng[tid];
        bs[tid] = lnb[tid];
    }
    __syncthreads();

    // per-column LN params, constant across all timesteps -> registers
    float gv[8], bv[8];
#pragma unroll
    for (int cc = 0; cc < 8; ++cc) { gv[cc] = gs[colb + cc]; bv[cc] = bs[colb + cc]; }

    for (int t = 0; t < SEQ; ++t) {
        __syncthreads();                       // hb/prev readers settled

        const size_t base = ((size_t)(bb * SEQ + t)) * DIM;
        qs[tid]  = q[base + tid];
        ks[tid]  = k[base + tid];
        vs[tid]  = v[base + tid];
        is_[tid] = ig[base + tid];
        fs[tid]  = fg[base + tid];
        __syncthreads();

        // per-column gate values for this step -> registers (b128 LDS loads)
        float kv[8], iv[8], fv[8];
#pragma unroll
        for (int cc = 0; cc < 8; ++cc) {
            kv[cc] = ks[colb + cc];
            iv[cc] = is_[colb + cc];
            fv[cc] = fs[colb + cc];
        }

        // ---- row phase: wave w owns rows m ≡ w (mod 8) for the whole scan
        float aacc[8];
#pragma unroll
        for (int cc = 0; cc < 8; ++cc) aacc[cc] = 0.0f;

        for (int rr = 0; rr < 32; ++rr) {
            const int m  = rr * 8 + wave;
            const float qm = qs[m];            // LDS broadcast
            const float vm = vs[m];
            float* crow = C + m * DIM + colb;  // 32B-aligned, 16B/lane pattern
            const float4 c0 = *(const float4*)(crow);
            const float4 c1 = *(const float4*)(crow + 4);
            float c[8] = { c0.x, c0.y, c0.z, c0.w, c1.x, c1.y, c1.z, c1.w };

            float s = 0.0f, s2 = 0.0f;
#pragma unroll
            for (int cc = 0; cc < 8; ++cc) { s += c[cc]; s2 += c[cc] * c[cc]; }
#pragma unroll
            for (int off = 16; off > 0; off >>= 1) {   // wave32 reduction
                s  += __shfl_xor(s,  off, 32);
                s2 += __shfl_xor(s2, off, 32);
            }
            const float mu   = s * (1.0f / 256.0f);
            const float var  = s2 * (1.0f / 256.0f) - mu * mu;
            const float rstd = rsqrtf(var + 1e-5f);

            float nw[8];
#pragma unroll
            for (int cc = 0; cc < 8; ++cc) {
                const float cn = (c[cc] - mu) * rstd * gv[cc] + bv[cc];
                nw[cc] = fv[cc] * cn + iv[cc] * (vm * kv[cc]);
                aacc[cc] += qm * nw[cc];
            }
            *(float4*)(crow)     = make_float4(nw[0], nw[1], nw[2], nw[3]);
            *(float4*)(crow + 4) = make_float4(nw[4], nw[5], nw[6], nw[7]);
        }
        {
            float* aprow = ap + wave * DIM + colb;
            *(float4*)(aprow)     = make_float4(aacc[0], aacc[1], aacc[2], aacc[3]);
            *(float4*)(aprow + 4) = make_float4(aacc[4], aacc[5], aacc[6], aacc[7]);
        }
        __syncthreads();

        // ---- column phase: attn combine + output-gate matvec
        float at = 0.0f;
#pragma unroll
        for (int w = 0; w < 8; ++w) at += ap[w * DIM + tid];

        float acc = bo[tid];
#pragma unroll 8
        for (int r = 0; r < DIM; ++r)
            acc += hb[r] * Wo[r * DIM + tid];  // hb broadcast; Wo coalesced (L2)
        const float o  = 1.0f / (1.0f + expf(-acc));
        const float hn = o * at;
        __syncthreads();                       // everyone done reading hb

        hb[tid] = hn;
        ys[base + tid] = hn;
    }

    // ---- epilogue: final hidden state + final C state (vectorized)
    __syncthreads();
    hf[bb * DIM + tid] = hb[tid];
    {
        const float4* C4  = (const float4*)C;
        float4*       Cf4 = (float4*)(Cf + (size_t)bb * 65536);
        for (int i = tid; i < 16384; i += 256) Cf4[i] = C4[i];
    }
}

// ---------------------------------------------------------------------------
extern "C" void kernel_launch(void* const* d_in, const int* in_sizes, int n_in,
                              void* d_out, int out_size, void* d_ws, size_t ws_size,
                              hipStream_t stream) {
    const float* X   = (const float*)d_in[0];
    const float* Wq  = (const float*)d_in[1];
    const float* bq  = (const float*)d_in[2];
    const float* Wk  = (const float*)d_in[3];
    const float* bk  = (const float*)d_in[4];
    const float* Wv  = (const float*)d_in[5];
    const float* bv  = (const float*)d_in[6];
    const float* Wi  = (const float*)d_in[7];
    const float* bi  = (const float*)d_in[8];
    const float* Wf  = (const float*)d_in[9];
    const float* bf_ = (const float*)d_in[10];
    const float* Wo  = (const float*)d_in[11];
    const float* bo  = (const float*)d_in[12];
    const float* lng = (const float*)d_in[13];
    const float* lnb = (const float*)d_in[14];

    float* ws = (float*)d_ws;                  // 5 x [2048,256] projections (10 MB)
    float* qp = ws + 0 * (size_t)NVEC;
    float* kp = ws + 1 * (size_t)NVEC;
    float* vp = ws + 2 * (size_t)NVEC;
    float* ip = ws + 3 * (size_t)NVEC;
    float* fp = ws + 4 * (size_t)NVEC;

    float* out = (float*)d_out;
    float* ys  = out;                          // [8,256,256]
    float* hf  = out + (size_t)BATCH * SEQ * DIM;          // [8,256]
    float* Cf  = hf + (size_t)BATCH * DIM;                 // [8,256,256]

    // Phase 1: 10240 tile-waves -> 1280 blocks of 256 threads
    mlstm_proj_kernel<<<dim3(1280), dim3(256), 0, stream>>>(
        X, Wq, bq, Wk, bk, Wv, bv, Wi, bi, Wf, bf_, ws);

    // Phase 2: one WGP per batch element, C resident in 272KB of LDS
    (void)hipFuncSetAttribute(reinterpret_cast<const void*>(mlstm_scan_kernel),
                              hipFuncAttributeMaxDynamicSharedMemorySize, SM_BYTES);
    mlstm_scan_kernel<<<dim3(BATCH), dim3(256), SM_BYTES, stream>>>(
        qp, kp, vp, ip, fp, Wo, bo, lng, lnb, ys, hf, Cf);
}
